// DynamicPrototypeRouter_87883620810888
// MI455X (gfx1250) — matrix operand order repaired and checked by reading, not compiled, not run on hardware
//
#include <hip/hip_runtime.h>
#include <math.h>

#define DEV __device__ __forceinline__

typedef __bf16 bf16;
typedef bf16 v16bf __attribute__((ext_vector_type(16)));
typedef bf16 v8bf  __attribute__((ext_vector_type(8)));
typedef float v8f  __attribute__((ext_vector_type(8)));
typedef unsigned long long ull;

static constexpr int kB = 16, kN = 4096, kD = 768, kM = 64, kR = 256;
static constexpr float kScale = 0.0625f;   // 256^-0.5

// ---------------------------------------------------------------------------
// WMMA fragment loaders (per ISA 7.12.2 layouts, bf16 16x16x32)
// A (16x32 MxK): lane L: row = L&15; half = L>>4; e0..7 -> k = 8*half + e;
//                e8..15 -> k = 16 + 8*half + (e-8).  => two b128 loads.
// B (32x16 KxN) from W stored row-major W[n][k]: lane L: col n = L&15;
//                same k order as A. => two b128 loads.
// ---------------------------------------------------------------------------
DEV v16bf frag_a(const bf16* tile, int ld, int lane) {
  int half = lane >> 4, r = lane & 15;
  const bf16* p = tile + (size_t)r * ld + 8 * half;
  v8bf lo = *(const v8bf*)p;
  v8bf hi = *(const v8bf*)(p + 16);
  return __builtin_shufflevector(lo, hi, 0,1,2,3,4,5,6,7,8,9,10,11,12,13,14,15);
}
DEV v16bf frag_b(const bf16* tile, int ld, int lane) {
  int half = lane >> 4, r = lane & 15;
  const bf16* p = tile + (size_t)r * ld + 8 * half;
  v8bf lo = *(const v8bf*)p;
  v8bf hi = *(const v8bf*)(p + 16);
  return __builtin_shufflevector(lo, hi, 0,1,2,3,4,5,6,7,8,9,10,11,12,13,14,15);
}

// ---------------------------------------------------------------------------
// Generic batched GEMM: out[row,col] = sum_k A[row,k] * W[col,k]
// Each wave computes 32 rows x (16*TN) cols via 2 x TN wmma_f32_16x16x32_bf16
// fragments (8 independent accumulators for TN=4 hide the WMMA->WMMA hazard;
// 8 WMMA per 12 b128 loads per k-step).
// Prefetch is unconditional + per-lane: speculative prefetch past the K range
// is silently dropped by hardware (ISA 7.3), and 32 lanes cover all rows of
// the next A/B k-slice.
// EPI: 0 = f32 store | 1 = +bias, exact GELU, bf16 store | 2 = bf16 store
//      3 = bf16 store transposed per batch: out[b, col, n] with row=(b<<12)|n
// ---------------------------------------------------------------------------
template<int WM, int WN, int TN, int EPI>
__global__ __launch_bounds__(WM*WN*32)
void gemm_bf16(const bf16* __restrict__ A, ull sA, int lda,
               const bf16* __restrict__ W, ull sW, int ldw,
               void* __restrict__ OutV, ull sO, int ldo,
               int K, const float* __restrict__ bias)
{
  const bf16* Ab = A + (size_t)blockIdx.z * sA;
  const bf16* Wb = W + (size_t)blockIdx.z * sW;
  const int lane = threadIdx.x & 31;
  const int wave = threadIdx.x >> 5;
  const int wm = wave % WM, wn = wave / WM;
  const int row0 = (blockIdx.y * WM + wm) * 32;
  const int col0 = (blockIdx.x * WN + wn) * (16 * TN);

  v8f zero = {0.f,0.f,0.f,0.f,0.f,0.f,0.f,0.f};
  v8f acc[2][TN];
  #pragma unroll
  for (int i = 0; i < 2; i++)
    #pragma unroll
    for (int j = 0; j < TN; j++) acc[i][j] = zero;

  const bf16* Ar[2];
  Ar[0] = Ab + (size_t)row0 * lda;
  Ar[1] = Ab + (size_t)(row0 + 16) * lda;
  const bf16* Wc[TN];
  #pragma unroll
  for (int j = 0; j < TN; j++) Wc[j] = Wb + (size_t)(col0 + 16 * j) * ldw;

  // per-lane prefetch bases: lane covers one row of the next k-slice
  const bf16* pfA = Ar[0] + (size_t)lane * lda;
  const bf16* pfW0 = Wc[0] + (size_t)lane * ldw;
  const bf16* pfW1 = Wc[0] + (size_t)(lane + 32) * ldw;   // rows 32..63 (TN=4)

  for (int k = 0; k < K; k += 32) {
    __builtin_prefetch(pfA + k + 32, 0, 1);               // global_prefetch_b8
    __builtin_prefetch(pfW0 + k + 32, 0, 1);
    if constexpr (TN == 4) __builtin_prefetch(pfW1 + k + 32, 0, 1);
    v16bf a[2], b[TN];
    #pragma unroll
    for (int i = 0; i < 2; i++) a[i] = frag_a(Ar[i] + k, lda, lane);
    #pragma unroll
    for (int j = 0; j < TN; j++) b[j] = frag_b(Wc[j] + k, ldw, lane);
    #pragma unroll
    for (int i = 0; i < 2; i++)
      #pragma unroll
      for (int j = 0; j < TN; j++)
        acc[i][j] = __builtin_amdgcn_wmma_f32_16x16x32_bf16(
            false, a[i], false, b[j], (short)0, acc[i][j], false, false);
  }

  const int half = lane >> 4, r = lane & 15;
  #pragma unroll
  for (int i = 0; i < 2; i++)
    #pragma unroll
    for (int j = 0; j < TN; j++)
      #pragma unroll
      for (int e = 0; e < 8; e++) {
        int row = row0 + 16*i + 8*half + e;
        int col = col0 + 16*j + r;
        float v = acc[i][j][e];
        if constexpr (EPI == 0) {
          float* O = (float*)OutV + (size_t)blockIdx.z * sO;
          O[(size_t)row * ldo + col] = v;
        } else if constexpr (EPI == 1) {
          bf16* O = (bf16*)OutV + (size_t)blockIdx.z * sO;
          float t = v + bias[col];
          t = 0.5f * t * (1.0f + erff(t * 0.70710678118654752f));
          O[(size_t)row * ldo + col] = (bf16)t;
        } else if constexpr (EPI == 2) {
          bf16* O = (bf16*)OutV + (size_t)blockIdx.z * sO;
          O[(size_t)row * ldo + col] = (bf16)v;
        } else {
          bf16* O = (bf16*)OutV;
          int bb = row >> 12, n = row & (kN - 1);
          O[((size_t)bb * kD + col) * kN + n] = (bf16)v;
        }
      }
}

// ---------------------------------------------------------------------------
// Block reductions
// ---------------------------------------------------------------------------
DEV float block_sum(float v, float* sh) {
  int t = threadIdx.x, n = blockDim.x;
  sh[t] = v; __syncthreads();
  for (int s = n >> 1; s > 0; s >>= 1) { if (t < s) sh[t] += sh[t + s]; __syncthreads(); }
  float r = sh[0]; __syncthreads(); return r;
}
DEV float block_max(float v, float* sh) {
  int t = threadIdx.x, n = blockDim.x;
  sh[t] = v; __syncthreads();
  for (int s = n >> 1; s > 0; s >>= 1) { if (t < s) sh[t] = fmaxf(sh[t], sh[t + s]); __syncthreads(); }
  float r = sh[0]; __syncthreads(); return r;
}

// ---------------------------------------------------------------------------
// f32 -> bf16 weight conversion
// ---------------------------------------------------------------------------
__global__ void f32_to_bf16_kernel(const float* __restrict__ in, bf16* __restrict__ out, int n) {
  int i = blockIdx.x * blockDim.x + threadIdx.x;
  if (i < n) out[i] = (bf16)in[i];
}

// LayerNorm(x) -> xn_bf16 ; x -> x_bf16 ; x -> xT_bf16[b][c][n]
__global__ __launch_bounds__(256)
void ln_convert_kernel(const float* __restrict__ x, const float* __restrict__ lw,
                       const float* __restrict__ lb, bf16* __restrict__ xn,
                       bf16* __restrict__ xb, bf16* __restrict__ xT)
{
  __shared__ float sh[256];
  int row = blockIdx.x, t = threadIdx.x;
  int b = row >> 12, n = row & (kN - 1);
  const float* xr = x + (size_t)row * kD;
  float v0 = xr[t], v1 = xr[t + 256], v2 = xr[t + 512];
  float mu = block_sum(v0 + v1 + v2, sh) * (1.f / kD);
  float d0 = v0 - mu, d1 = v1 - mu, d2 = v2 - mu;
  float var = block_sum(d0*d0 + d1*d1 + d2*d2, sh) * (1.f / kD);
  float rs = rsqrtf(var + 1e-5f);
  float vv[3] = {v0, v1, v2};
  #pragma unroll
  for (int ii = 0; ii < 3; ii++) {
    int c = t + ii * 256;
    float z = (vv[ii] - mu) * rs * lw[c] + lb[c];
    xn[(size_t)row * kD + c] = (bf16)z;
    xb[(size_t)row * kD + c] = (bf16)vv[ii];
    xT[((size_t)b * kD + c) * kN + n] = (bf16)vv[ii];
  }
}

// softmax over M=64, write transposed assignT[b][m][n]
__global__ __launch_bounds__(64)
void assign_softmax_kernel(const float* __restrict__ logits, bf16* __restrict__ aT)
{
  __shared__ float sh[64];
  int row = blockIdx.x, t = threadIdx.x;
  int b = row >> 12, n = row & (kN - 1);
  float l = logits[(size_t)row * kM + t];
  float mx = block_max(l, sh);
  float e = expf(l - mx);
  float s = block_sum(e, sh);
  aT[((size_t)b * kM + t) * kN + n] = (bf16)(e / s);
}

// pw[b,m] = max(sum_n assignT, 1e-6)
__global__ __launch_bounds__(256)
void pw_kernel(const bf16* __restrict__ aT, float* __restrict__ pw)
{
  __shared__ float sh[256];
  int bm = blockIdx.x, t = threadIdx.x;
  const bf16* r = aT + (size_t)bm * kN;
  float s = 0.f;
  for (int i = t; i < kN; i += 256) s += (float)r[i];
  s = block_sum(s, sh);
  if (t == 0) pw[bm] = fmaxf(s, 1e-6f);
}

// pc = l2n(LN(pc_raw / pw)); qs = pc + proto_q
__global__ __launch_bounds__(256)
void pc_norm_kernel(const float* __restrict__ pc_raw, const float* __restrict__ pw,
                    const float* __restrict__ nw, const float* __restrict__ nb,
                    const float* __restrict__ proto_q,
                    float* __restrict__ pcf, bf16* __restrict__ qs)
{
  __shared__ float sh[256];
  int bm = blockIdx.x, t = threadIdx.x, m = bm & (kM - 1);
  const float* src = pc_raw + (size_t)bm * kD;
  float iw = 1.f / pw[bm];
  float v[3];
  #pragma unroll
  for (int ii = 0; ii < 3; ii++) v[ii] = src[t + ii * 256] * iw;
  float mu = block_sum(v[0] + v[1] + v[2], sh) * (1.f / kD);
  float var = block_sum((v[0]-mu)*(v[0]-mu) + (v[1]-mu)*(v[1]-mu) + (v[2]-mu)*(v[2]-mu), sh) * (1.f / kD);
  float rs = rsqrtf(var + 1e-5f);
  float z[3];
  #pragma unroll
  for (int ii = 0; ii < 3; ii++) { int c = t + ii*256; z[ii] = (v[ii]-mu)*rs*nw[c] + nb[c]; }
  float nrm = sqrtf(block_sum(z[0]*z[0] + z[1]*z[1] + z[2]*z[2], sh));
  float inv = 1.f / fmaxf(nrm, 1e-12f);
  #pragma unroll
  for (int ii = 0; ii < 3; ii++) {
    int c = t + ii * 256;
    float pv = z[ii] * inv;
    pcf[(size_t)bm * kD + c] = pv;
    qs[(size_t)bm * kD + c] = (bf16)(pv + proto_q[(size_t)m * kD + c]);
  }
}

// softmax over N=4096 of attn logits * SCALE -> bf16
__global__ __launch_bounds__(256)
void attn_softmax_kernel(const float* __restrict__ logits, bf16* __restrict__ attn)
{
  __shared__ float sh[256];
  size_t base = (size_t)blockIdx.x * kN;
  int t = threadIdx.x;
  float l[16]; float mx = -1e30f;
  #pragma unroll
  for (int i = 0; i < 16; i++) { l[i] = logits[base + t + i*256] * kScale; mx = fmaxf(mx, l[i]); }
  mx = block_max(mx, sh);
  float s = 0.f;
  #pragma unroll
  for (int i = 0; i < 16; i++) { l[i] = expf(l[i] - mx); s += l[i]; }
  s = block_sum(s, sh);
  float inv = 1.f / s;
  #pragma unroll
  for (int i = 0; i < 16; i++) attn[base + t + i*256] = (bf16)(l[i] * inv);
}

// prototypes = l2n(LN(pc + sigmoid(gate)*pr)) -> f32 out + bf16 ws
__global__ __launch_bounds__(256)
void proto_kernel(const float* __restrict__ pcf, const float* __restrict__ pr,
                  const float* __restrict__ gate, const float* __restrict__ nw,
                  const float* __restrict__ nb, float* __restrict__ proto_out,
                  bf16* __restrict__ proto_bf)
{
  __shared__ float sh[256];
  int bm = blockIdx.x, t = threadIdx.x;
  float g = 1.f / (1.f + expf(-gate[0]));
  float v[3];
  #pragma unroll
  for (int ii = 0; ii < 3; ii++) {
    int c = t + ii * 256;
    v[ii] = pcf[(size_t)bm * kD + c] + g * pr[(size_t)bm * kD + c];
  }
  float mu = block_sum(v[0] + v[1] + v[2], sh) * (1.f / kD);
  float var = block_sum((v[0]-mu)*(v[0]-mu) + (v[1]-mu)*(v[1]-mu) + (v[2]-mu)*(v[2]-mu), sh) * (1.f / kD);
  float rs = rsqrtf(var + 1e-5f);
  float z[3];
  #pragma unroll
  for (int ii = 0; ii < 3; ii++) { int c = t + ii*256; z[ii] = (v[ii]-mu)*rs*nw[c] + nb[c]; }
  float nrm = sqrtf(block_sum(z[0]*z[0] + z[1]*z[1] + z[2]*z[2], sh));
  float inv = 1.f / fmaxf(nrm, 1e-12f);
  #pragma unroll
  for (int ii = 0; ii < 3; ii++) {
    int c = t + ii * 256;
    float pv = z[ii] * inv;
    proto_out[(size_t)bm * kD + c] = pv;
    proto_bf[(size_t)bm * kD + c] = (bf16)pv;
  }
}

// l2-normalize rows of length 256, f32 -> bf16
__global__ __launch_bounds__(256)
void l2n_kernel(const float* __restrict__ in, bf16* __restrict__ out)
{
  __shared__ float sh[256];
  size_t base = (size_t)blockIdx.x * kR;
  float v = in[base + threadIdx.x];
  float s = block_sum(v * v, sh);
  float inv = 1.f / fmaxf(sqrtf(s), 1e-12f);
  out[base + threadIdx.x] = (bf16)(v * inv);
}

// per token: softmax over 64 (implicit), x_score = max prob, belong = argmax
__global__ __launch_bounds__(64)
void score_kernel(const float* __restrict__ logits, float* __restrict__ xs,
                  int* __restrict__ belong, float* __restrict__ key)
{
  __shared__ float sh[64];
  __shared__ int si[64];
  int row = blockIdx.x, t = threadIdx.x;
  float l = logits[(size_t)row * kM + t] * kScale;
  float mx = block_max(l, sh);
  si[t] = (l == mx) ? t : 0x7fffffff;
  __syncthreads();
  for (int s2 = 32; s2 > 0; s2 >>= 1) { if (t < s2) si[t] = min(si[t], si[t + s2]); __syncthreads(); }
  int idx = si[0]; __syncthreads();
  float s = block_sum(expf(l - mx), sh);
  if (t == 0) {
    float p = 1.f / s;                 // exp(0)/sum
    xs[row] = p;
    belong[row] = idx;
    key[row] = (float)idx + 0.5f * (1.f - p);
  }
}

// per-batch bitonic argsort over N=4096 keys (ascending, idx tiebreak)
__global__ __launch_bounds__(1024)
void sort_kernel(const float* __restrict__ key, int* __restrict__ sidx)
{
  __shared__ float sk[kN];
  __shared__ int six[kN];
  int b = blockIdx.x, t = threadIdx.x;
  for (int i = t; i < kN; i += 1024) { sk[i] = key[(size_t)b * kN + i]; six[i] = i; }
  __syncthreads();
  for (int kk = 2; kk <= kN; kk <<= 1)
    for (int j = kk >> 1; j > 0; j >>= 1) {
      for (int i = t; i < kN; i += 1024) {
        int ixj = i ^ j;
        if (ixj > i) {
          float a = sk[i], c = sk[ixj];
          int ia = six[i], ic = six[ixj];
          bool up = ((i & kk) == 0);
          bool sw = up ? (a > c || (a == c && ia > ic))
                       : (a < c || (a == c && ia < ic));
          if (sw) { sk[i] = c; sk[ixj] = a; six[i] = ic; six[ixj] = ia; }
        }
      }
      __syncthreads();
    }
  for (int i = t; i < kN; i += 1024) sidx[(size_t)b * kN + i] = six[i];
}

// gather sorted outputs
__global__ __launch_bounds__(256)
void gather_kernel(const float* __restrict__ x, const int* __restrict__ sidx,
                   const int* __restrict__ belong, const float* __restrict__ xs,
                   float* __restrict__ sx, int* __restrict__ sidx_out,
                   int* __restrict__ sbelong, float* __restrict__ sscores)
{
  int row = blockIdx.x, t = threadIdx.x;
  int b = row >> 12;
  int i = sidx[row];
  const float* src = x + ((size_t)b * kN + i) * kD;
  float* dst = sx + (size_t)row * kD;
  dst[t] = src[t]; dst[t + 256] = src[t + 256]; dst[t + 512] = src[t + 512];
  if (t == 0) {
    sidx_out[row] = i;
    sbelong[row] = belong[(size_t)b * kN + i];
    sscores[row] = xs[(size_t)b * kN + i];
  }
}

// ---------------------------------------------------------------------------
extern "C" void kernel_launch(void* const* d_in, const int* in_sizes, int n_in,
                              void* d_out, int out_size, void* d_ws, size_t ws_size,
                              hipStream_t stream) {
  const float* x        = (const float*)d_in[0];
  const float* eln_w    = (const float*)d_in[1];
  const float* eln_b    = (const float*)d_in[2];
  const float* embed_w  = (const float*)d_in[3];
  const float* embed_b  = (const float*)d_in[4];
  const float* assign_w = (const float*)d_in[5];
  const float* proto_q  = (const float*)d_in[6];
  const float* rq_w     = (const float*)d_in[7];
  const float* rk_w     = (const float*)d_in[8];
  const float* rv_w     = (const float*)d_in[9];
  const float* tok_w    = (const float*)d_in[10];
  const float* pro_w    = (const float*)d_in[11];
  const float* gate     = (const float*)d_in[12];
  const float* pnorm_w  = (const float*)d_in[13];
  const float* pnorm_b  = (const float*)d_in[14];

  float* out = (float*)d_out;
  float* o_sx     = out;                                    // [B,N,D]
  int*   o_sidx   = (int*)(out + (size_t)kB * kN * kD);     // [B,N,1]
  int*   o_belong = o_sidx + kB * kN;                       // [B,N]
  float* o_scores = (float*)(o_belong + kB * kN);           // [B,N]
  float* o_proto  = o_scores + kB * kN;                     // [B,M,D]

  char* p = (char*)d_ws;
  auto alloc = [&](size_t bytes) -> char* {
    char* r = p; p += (bytes + 255) & ~(size_t)255; return r;
  };
  const size_t BN = (size_t)kB * kN;

  bf16* w_embed = (bf16*)alloc((size_t)kD * kD * 2);
  bf16* w_assign= (bf16*)alloc((size_t)kM * kD * 2);
  bf16* w_rq    = (bf16*)alloc((size_t)kR * kD * 2);
  bf16* w_rk    = (bf16*)alloc((size_t)kR * kD * 2);
  bf16* w_rv    = (bf16*)alloc((size_t)kD * kD * 2);
  bf16* w_tok   = (bf16*)alloc((size_t)kR * kD * 2);
  bf16* w_pro   = (bf16*)alloc((size_t)kR * kD * 2);

  bf16* xn      = (bf16*)alloc(BN * kD * 2);              // LN(x)
  bf16* xb      = (bf16*)alloc(BN * kD * 2);              // x as bf16
  bf16* xT      = (bf16*)alloc(BN * kD * 2);              // x transposed per batch
  bf16* emb     = (bf16*)alloc(BN * kD * 2);              // GELU embed
  bf16* vT      = (bf16*)alloc(BN * kD * 2);              // v transposed per batch
  bf16* k_bf    = (bf16*)alloc(BN * kR * 2);
  bf16* tf_bf   = (bf16*)alloc(BN * kR * 2);
  float* asg_lg = (float*)alloc(BN * kM * 4);
  bf16* assignT = (bf16*)alloc((size_t)kB * kM * kN * 2);
  float* pw     = (float*)alloc((size_t)kB * kM * 4);
  float* pc_raw = (float*)alloc((size_t)kB * kM * kD * 4);
  float* pc_f   = (float*)alloc((size_t)kB * kM * kD * 4);
  bf16* qs      = (bf16*)alloc((size_t)kB * kM * kD * 2);
  bf16* q_bf    = (bf16*)alloc((size_t)kB * kM * kR * 2);
  float* at_lg  = (float*)alloc((size_t)kB * kM * kN * 4);
  bf16* attn_bf = (bf16*)alloc((size_t)kB * kM * kN * 2);
  float* pr     = (float*)alloc((size_t)kB * kM * kD * 4);
  bf16* proto_bf= (bf16*)alloc((size_t)kB * kM * kD * 2);
  float* tf_lg  = (float*)alloc(BN * kR * 4);
  float* pf_lg  = (float*)alloc((size_t)kB * kM * kR * 4);
  bf16* pf_bf   = (bf16*)alloc((size_t)kB * kM * kR * 2);
  float* sc_lg  = (float*)alloc(BN * kM * 4);
  float* xs     = (float*)alloc(BN * 4);
  int*   belong = (int*)alloc(BN * 4);
  float* skey   = (float*)alloc(BN * 4);
  int*   sidx   = (int*)alloc(BN * 4);

  auto conv = [&](const float* src, bf16* dst, int n) {
    f32_to_bf16_kernel<<<(n + 255) / 256, 256, 0, stream>>>(src, dst, n);
  };
  conv(embed_w,  w_embed,  kD * kD);
  conv(assign_w, w_assign, kM * kD);
  conv(rq_w,     w_rq,     kR * kD);
  conv(rk_w,     w_rk,     kR * kD);
  conv(rv_w,     w_rv,     kD * kD);
  conv(tok_w,    w_tok,    kR * kD);
  conv(pro_w,    w_pro,    kR * kD);

  // LN + bf16 conversions of x
  ln_convert_kernel<<<kB * kN, 256, 0, stream>>>(x, eln_w, eln_b, xn, xb, xT);

  // embed = GELU(LN(x) @ embed_w^T + b)                    [65536 x 768]
  gemm_bf16<4,2,4,1><<<dim3(kD/128, (kB*kN)/128, 1), 256, 0, stream>>>(
      xn, 0, kD, w_embed, 0, kD, emb, 0, kD, kD, embed_b);

  // assignment logits = embed @ assign_w^T                 [65536 x 64]
  gemm_bf16<4,2,2,0><<<dim3(1, (kB*kN)/128, 1), 256, 0, stream>>>(
      emb, 0, kD, w_assign, 0, kD, asg_lg, 0, kM, kD, nullptr);
  assign_softmax_kernel<<<kB * kN, 64, 0, stream>>>(asg_lg, assignT);
  pw_kernel<<<kB * kM, 256, 0, stream>>>(assignT, pw);

  // pc_raw[b] = assignT[b] (64xN) x xT[b] (DxN)            [64 x 768] per b
  gemm_bf16<2,2,4,0><<<dim3(kD/128, 1, kB), 128, 0, stream>>>(
      assignT, (ull)kM*kN, kN, xT, (ull)kD*kN, kN, pc_raw, (ull)kM*kD, kD, kN, nullptr);
  pc_norm_kernel<<<kB * kM, 256, 0, stream>>>(pc_raw, pw, pnorm_w, pnorm_b, proto_q, pc_f, qs);

  // q = qs @ rq_w^T    [1024 x 256]
  gemm_bf16<2,2,4,2><<<dim3(kR/128, (kB*kM)/64, 1), 128, 0, stream>>>(
      qs, 0, kD, w_rq, 0, kD, q_bf, 0, kR, kD, nullptr);
  // k = embed @ rk_w^T [65536 x 256]
  gemm_bf16<4,2,4,2><<<dim3(kR/128, (kB*kN)/128, 1), 256, 0, stream>>>(
      emb, 0, kD, w_rk, 0, kD, k_bf, 0, kR, kD, nullptr);
  // v = x @ rv_w^T, stored transposed per batch vT[b][c][n]
  gemm_bf16<4,2,4,3><<<dim3(kD/128, (kB*kN)/128, 1), 256, 0, stream>>>(
      xb, 0, kD, w_rv, 0, kD, vT, 0, 0, kD, nullptr);

  // attn logits[b] = q[b] (64xR) x k[b] (NxR)^T            [64 x 4096]
  gemm_bf16<2,2,4,0><<<dim3(kN/128, 1, kB), 128, 0, stream>>>(
      q_bf, (ull)kM*kR, kR, k_bf, (ull)kN*kR, kR, at_lg, (ull)kM*kN, kN, kR, nullptr);
  attn_softmax_kernel<<<kB * kM, 256, 0, stream>>>(at_lg, attn_bf);

  // pr[b] = attn[b] (64xN) x vT[b] (DxN)                   [64 x 768]
  gemm_bf16<2,2,4,0><<<dim3(kD/128, 1, kB), 128, 0, stream>>>(
      attn_bf, (ull)kM*kN, kN, vT, (ull)kD*kN, kN, pr, (ull)kM*kD, kD, kN, nullptr);
  proto_kernel<<<kB * kM, 256, 0, stream>>>(pc_f, pr, gate, pnorm_w, pnorm_b, o_proto, proto_bf);

  // tf = l2n(embed @ tok_w^T)                              [65536 x 256]
  gemm_bf16<4,2,4,0><<<dim3(kR/128, (kB*kN)/128, 1), 256, 0, stream>>>(
      emb, 0, kD, w_tok, 0, kD, tf_lg, 0, kR, kD, nullptr);
  l2n_kernel<<<kB * kN, 256, 0, stream>>>(tf_lg, tf_bf);

  // pf = l2n(prototypes @ pro_w^T)                         [1024 x 256]
  gemm_bf16<2,2,4,0><<<dim3(kR/128, (kB*kM)/64, 1), 128, 0, stream>>>(
      proto_bf, 0, kD, w_pro, 0, kD, pf_lg, 0, kR, kD, nullptr);
  l2n_kernel<<<kB * kM, 256, 0, stream>>>(pf_lg, pf_bf);

  // scores logits[b] = tf[b] (NxR) x pf[b] (64xR)^T        [4096 x 64]
  gemm_bf16<4,2,2,0><<<dim3(1, kN/128, kB), 256, 0, stream>>>(
      tf_bf, (ull)kN*kR, kR, pf_bf, (ull)kM*kR, kR, sc_lg, (ull)kN*kM, kM, kR, nullptr);
  score_kernel<<<kB * kN, 64, 0, stream>>>(sc_lg, xs, belong, skey);

  sort_kernel<<<kB, 1024, 0, stream>>>(skey, sidx);
  gather_kernel<<<kB * kN, 256, 0, stream>>>(x, sidx, belong, xs,
                                             o_sx, o_sidx, o_belong, o_scores);
  (void)in_sizes; (void)n_in; (void)out_size; (void)ws_size;
}
